// lstm_seq2seq_7576322310973
// MI455X (gfx1250) — matrix operand order repaired and checked by reading, not compiled, not run on hardware
//
#include <hip/hip_runtime.h>

// ---------------- problem constants ----------------
#define HID   512
#define BATCH 512
#define SEQ   1000
#define TGT   1000
#define NSTEP (SEQ + TGT)

// ---------------- launch geometry ------------------
// 32 row-tiles x 32 col-tiles of 16x16. Block = 256 thr = 8 waves.
// Block covers 4 row-tiles x 2 col-tiles. Grid = 8 rowgroups x 16 colgroups.
#define NBLK 128

// ---------------- LDS layout -----------------------
// B tiles: [coltile 2][gate 4][kchunk 16][lane 32] * 32 bytes = 128 KB
#define SMEM_B    (2 * 4 * 16 * 32 * 32)
#define SMEM_XOFF SMEM_B            // 64 floats: per-block x staging
#define SMEM_POFF (SMEM_B + 256)    // 128 floats: pred partial staging
#define SMEM_TOTAL (SMEM_B + 256 + 512)

typedef __attribute__((ext_vector_type(16))) __bf16        v16bf;
typedef __attribute__((ext_vector_type(8)))  float         v8f;
typedef __attribute__((ext_vector_type(4)))  unsigned int  v4u;

union BPack { v4u u[2]; v16bf v; };

__device__ inline unsigned short f2bf(float f) {
  unsigned int u = __float_as_uint(f);
  u += 0x7FFFu + ((u >> 16) & 1u);           // round-to-nearest-even
  return (unsigned short)(u >> 16);
}

__device__ inline float sigm(float x) { return 1.0f / (1.0f + __expf(-x)); }

// Sense-reversal device-wide barrier (one per LSTM step).
__device__ inline void grid_sync(int* cnt, int* gen) {
  __threadfence();
  __syncthreads();
  if (threadIdx.x == 0) {
    int g = __hip_atomic_load(gen, __ATOMIC_RELAXED, __HIP_MEMORY_SCOPE_AGENT);
    int a = __hip_atomic_fetch_add(cnt, 1, __ATOMIC_ACQ_REL, __HIP_MEMORY_SCOPE_AGENT);
    if (a == NBLK - 1) {
      __hip_atomic_store(cnt, 0, __ATOMIC_RELAXED, __HIP_MEMORY_SCOPE_AGENT);
      __hip_atomic_fetch_add(gen, 1, __ATOMIC_RELEASE, __HIP_MEMORY_SCOPE_AGENT);
    } else {
      while (__hip_atomic_load(gen, __ATOMIC_ACQUIRE, __HIP_MEMORY_SCOPE_AGENT) == g)
        __builtin_amdgcn_s_sleep(2);
    }
  }
  __syncthreads();
  __threadfence();
}

// Stage this block's Whh slice (2 coltiles x 4 gates x K=512, bf16) into LDS,
// pre-swizzled into this kernel's per-lane B-operand layout (contiguous K, same as A):
//   B lane l: column n = l&15, K = kc*32 + (l>>4)*16 + [0..15]  (32 contiguous B)
__device__ inline void fill_B_lds(char* smem, const unsigned short* whh, int wgCol) {
  for (int i = threadIdx.x; i < 2 * 4 * 16 * 32; i += blockDim.x) {
    int l  =  i        & 31;
    int kc = (i >> 5)  & 15;
    int g  = (i >> 9)  & 3;
    int cc = (i >> 11) & 1;
    int j  = g * HID + wgCol * 32 + cc * 16 + (l & 15);   // row of Whh
    int k0 = kc * 32 + ((l >> 4) << 4);
    const v4u* src = (const v4u*)(whh + j * HID + k0);
    v4u lo = src[0];
    v4u hi = src[1];
    v4u* dst = (v4u*)(smem + i * 32);
    dst[0] = lo;
    dst[1] = hi;
  }
}

__global__ void __launch_bounds__(256, 1)
lstm_seq2seq_prep(const float* encWhh, const float* decWhh,
                  const float* encBih, const float* encBhh,
                  const float* decBih, const float* decBhh,
                  unsigned short* whhEnc, unsigned short* whhDec,
                  float* bEnc, float* bDec,
                  unsigned short* h0, unsigned short* h1,
                  float* predPart, int* sync) {
  int tid = blockIdx.x * blockDim.x + threadIdx.x;
  int stride = gridDim.x * blockDim.x;
  for (int i = tid; i < 4 * HID * HID; i += stride) {
    whhEnc[i] = f2bf(encWhh[i]);
    whhDec[i] = f2bf(decWhh[i]);
  }
  for (int i = tid; i < 4 * HID; i += stride) {
    bEnc[i] = encBih[i] + encBhh[i];
    bDec[i] = decBih[i] + decBhh[i];
  }
  for (int i = tid; i < BATCH * HID; i += stride) { h0[i] = 0; h1[i] = 0; }
  for (int i = tid; i < 16 * BATCH; i += stride) predPart[i] = 0.0f;
  if (tid == 0) { sync[0] = 0; sync[1] = 0; }
}

__global__ void __launch_bounds__(256, 1)
lstm_seq2seq_main(const float* __restrict__ input,
                  const float* __restrict__ encWih, const float* __restrict__ decWih,
                  const float* __restrict__ linW,   const float* __restrict__ linB,
                  const unsigned short* __restrict__ whhEnc,
                  const unsigned short* __restrict__ whhDec,
                  const float* __restrict__ bEnc, const float* __restrict__ bDec,
                  unsigned short* h0, unsigned short* h1,
                  float* predPart, int* sync, float* out) {
  extern __shared__ char smem[];
  float* xS = (float*)(smem + SMEM_XOFF);   // 64 rows of x for this block
  float* pS = (float*)(smem + SMEM_POFF);   // 8 waves x 16 local rows

  const int wgRow = blockIdx.x >> 4;        // 0..7  (64 batch rows each)
  const int wgCol = blockIdx.x & 15;        // 0..15 (32 hidden cols each)
  const int tid   = threadIdx.x;
  const int wave  = tid >> 5;
  const int lane  = tid & 31;
  const int rowTile = wave >> 1;            // 0..3
  const int colTile = wave & 1;             // 0..1
  const int rowBase = wgRow * 64 + rowTile * 16;
  const int colBase = wgCol * 32 + colTile * 16;

  const int nHalf = lane >> 4;              // 0 or 1 (C/D upper half -> M += 8)
  const int colN  = colBase + (lane & 15);  // hidden index 0..511
  const int mRow  = rowBase + (lane & 15);  // A-operand row for this lane
  const int aHalf = nHalf << 4;             // A K-offset (0 or 16), contiguous K per lane half
  const char* bWave = smem + (size_t)(colTile * 4) * 16 * 1024;

  float c[8];
  #pragma unroll
  for (int v = 0; v < 8; ++v) c[v] = 0.0f;

  int* cnt = sync;
  int* gen = sync + 1;

  for (int t = 0; t < NSTEP; ++t) {
    grid_sync(cnt, gen);                    // step t-1 fully visible

    const bool dec = (t >= SEQ);
    if (t == 0) {
      fill_B_lds(smem, whhEnc, wgCol);
    } else if (t == SEQ) {
      __syncthreads();
      fill_B_lds(smem, whhDec, wgCol);
    }

    // ---- stage x for this block's 64 rows (and emit pred of step t-1) ----
    if (tid < 64) {
      int row = wgRow * 64 + tid;
      float xv;
      if (!dec) {
        xv = input[t * BATCH + row];
      } else if (t == SEQ) {
        xv = 0.0f;
      } else {
        float s = linB[0];
        #pragma unroll
        for (int cg = 0; cg < 16; ++cg) s += predPart[cg * BATCH + row];
        xv = s;
        if (wgCol == 0) out[(t - SEQ - 1) * BATCH + row] = s;
      }
      xS[tid] = xv;
    }
    __syncthreads();

    const unsigned short* hR = (t & 1) ? h1 : h0;
    unsigned short*       hW = (t & 1) ? h0 : h1;
    const float* bias = dec ? bDec : bEnc;
    const float* wih  = dec ? decWih : encWih;

    // ---- gates = h @ Whh^T via bf16 WMMA, K = 512 in chunks of 32 ----
    v8f zero = {0.f, 0.f, 0.f, 0.f, 0.f, 0.f, 0.f, 0.f};
    v8f acc[4];
    #pragma unroll
    for (int g = 0; g < 4; ++g) acc[g] = zero;

    #pragma unroll 4
    for (int kc = 0; kc < 16; ++kc) {
      BPack A;
      const unsigned short* ap = hR + mRow * HID + kc * 32 + aHalf;
      A.u[0] = *(const v4u*)ap;
      A.u[1] = *(const v4u*)(ap + 8);
      #pragma unroll
      for (int g = 0; g < 4; ++g) {
        BPack B;
        const v4u* bp = (const v4u*)(bWave + (size_t)(g * 16 + kc) * 1024 + lane * 32);
        B.u[0] = bp[0];
        B.u[1] = bp[1];
        acc[g] = __builtin_amdgcn_wmma_f32_16x16x32_bf16(
            false, A.v, false, B.v, (short)0, acc[g], false, false);
      }
    }

    // ---- elementwise LSTM cell (c lives in registers) ----
    float bi = bias[0 * HID + colN], bf_ = bias[1 * HID + colN];
    float bg = bias[2 * HID + colN], bo  = bias[3 * HID + colN];
    float wi = wih[0 * HID + colN],  wf  = wih[1 * HID + colN];
    float wg = wih[2 * HID + colN],  wo  = wih[3 * HID + colN];
    float lw = linW[colN];

    #pragma unroll
    for (int v = 0; v < 8; ++v) {
      int lrow = rowTile * 16 + nHalf * 8 + v;      // 0..63 within block
      int row  = wgRow * 64 + lrow;
      float xv = xS[lrow];
      float gi = acc[0][v] + bi  + xv * wi;
      float gf = acc[1][v] + bf_ + xv * wf;
      float gg = acc[2][v] + bg  + xv * wg;
      float go = acc[3][v] + bo  + xv * wo;
      float cn = sigm(gf) * c[v] + sigm(gi) * tanhf(gg);
      float hn = sigm(go) * tanhf(cn);
      c[v] = cn;
      hW[row * HID + colN] = f2bf(hn);

      // partial pred: reduce over the 16 lanes holding this row's columns
      float p = hn * lw;
      p += __shfl_xor(p, 1, 32);
      p += __shfl_xor(p, 2, 32);
      p += __shfl_xor(p, 4, 32);
      p += __shfl_xor(p, 8, 32);
      if ((lane & 15) == 0) pS[wave * 16 + nHalf * 8 + v] = p;
    }
    __syncthreads();

    if (dec && tid < 64) {
      int rt = tid >> 4, lr = tid & 15;
      float s = pS[(rt * 2 + 0) * 16 + lr] + pS[(rt * 2 + 1) * 16 + lr];
      predPart[wgCol * BATCH + wgRow * 64 + tid] = s;
    }
  }

  // ---- emit final decoder step's prediction ----
  grid_sync(cnt, gen);
  if (wgCol == 0 && tid < 64) {
    int row = wgRow * 64 + tid;
    float s = linB[0];
    #pragma unroll
    for (int cg = 0; cg < 16; ++cg) s += predPart[cg * BATCH + row];
    out[(TGT - 1) * BATCH + row] = s;
  }
}

extern "C" void kernel_launch(void* const* d_in, const int* in_sizes, int n_in,
                              void* d_out, int out_size, void* d_ws, size_t ws_size,
                              hipStream_t stream) {
  const float* input  = (const float*)d_in[0];
  const float* encWih = (const float*)d_in[1];
  const float* encWhh = (const float*)d_in[2];
  const float* encBih = (const float*)d_in[3];
  const float* encBhh = (const float*)d_in[4];
  const float* decWih = (const float*)d_in[5];
  const float* decWhh = (const float*)d_in[6];
  const float* decBih = (const float*)d_in[7];
  const float* decBhh = (const float*)d_in[8];
  const float* linW   = (const float*)d_in[9];
  const float* linB   = (const float*)d_in[10];
  float* out = (float*)d_out;

  // workspace layout (bytes)
  char* ws = (char*)d_ws;
  unsigned short* h0      = (unsigned short*)(ws);                 // 512 KB
  unsigned short* h1      = (unsigned short*)(ws + 524288);        // 512 KB
  float*          predPart= (float*)(ws + 1048576);                // 32 KB
  float*          bEnc    = (float*)(ws + 1081344);                // 8 KB
  float*          bDec    = (float*)(ws + 1089536);                // 8 KB
  unsigned short* whhEnc  = (unsigned short*)(ws + 1097728);       // 2 MB
  unsigned short* whhDec  = (unsigned short*)(ws + 3194880);       // 2 MB
  int*            sync    = (int*)(ws + 5292032);                  // 2 ints

  (void)hipFuncSetAttribute((const void*)lstm_seq2seq_main,
                            hipFuncAttributeMaxDynamicSharedMemorySize,
                            SMEM_TOTAL);

  lstm_seq2seq_prep<<<256, 256, 0, stream>>>(
      encWhh, decWhh, encBih, encBhh, decBih, decBhh,
      whhEnc, whhDec, bEnc, bDec, h0, h1, predPart, sync);

  lstm_seq2seq_main<<<NBLK, 256, SMEM_TOTAL, stream>>>(
      input, encWih, decWih, linW, linB,
      whhEnc, whhDec, bEnc, bDec, h0, h1, predPart, sync, out);
}
